// MultiDirectionalGridAttention_19542101196881
// MI455X (gfx1250) — matrix operand order repaired and checked by reading, not compiled, not run on hardware
//
#include <hip/hip_runtime.h>

// ---------------------------------------------------------------------------
// Types for CDNA5 WMMA (gfx1250, wave32)
// ---------------------------------------------------------------------------
typedef __bf16 bf16;
typedef __attribute__((ext_vector_type(4)))  __bf16 v4bf;
typedef __attribute__((ext_vector_type(8)))  __bf16 v8bf;
typedef __attribute__((ext_vector_type(16))) __bf16 v16bf;
typedef __attribute__((ext_vector_type(8)))  float  v8f;
typedef int v4i __attribute__((vector_size(4 * sizeof(int))));

__device__ __forceinline__ v16bf pack16(v8bf lo, v8bf hi) {
  v16bf r;
#pragma unroll
  for (int i = 0; i < 8; ++i) { r[i] = lo[i]; r[i + 8] = hi[i]; }
  return r;
}

// Async global->LDS 16-byte copy (CDNA5 GLOBAL_LOAD_ASYNC_TO_LDS_B128,
// tracked with ASYNCcnt). Per active lane: LDS[l] = MEM[g], 16 bytes.
__device__ __forceinline__ void async_cp16(const bf16* g, bf16* l) {
  __builtin_amdgcn_global_load_async_to_lds_b128(
      (__attribute__((address_space(1))) v4i*)(uintptr_t)g,
      (__attribute__((address_space(3))) v4i*)l,
      /*imm offset*/ 0, /*cpol*/ 0);
}

__device__ __forceinline__ void wait_asynccnt0() {
  asm volatile("s_wait_asynccnt 0x0" ::: "memory");
}

// ---------------------------------------------------------------------------
// fp32 -> bf16 conversion (vectorized, grid-stride)
// ---------------------------------------------------------------------------
__global__ void cvt_f32_bf16(const float* __restrict__ s, bf16* __restrict__ d, long n4) {
  long i  = (long)blockIdx.x * blockDim.x + threadIdx.x;
  long st = (long)gridDim.x * blockDim.x;
  for (; i < n4; i += st) {
    float4 v = ((const float4*)s)[i];
    v4bf o;
    o[0] = (bf16)v.x; o[1] = (bf16)v.y; o[2] = (bf16)v.z; o[3] = (bf16)v.w;
    ((v4bf*)d)[i] = o;
  }
}

// ---------------------------------------------------------------------------
// bf16 WMMA GEMM with double-buffered async global->LDS staging:
//   C[M,N] = A[M,K] @ W[N,K]^T   (row-major)
// Block = 256 threads = 8 waves (wave32). Block tile 128x128, K-step 32.
// Tiles staged via GLOBAL_LOAD_ASYNC_TO_LDS_B128 (ASYNCcnt), double buffer.
// Waves arranged 4(M) x 2(N); each wave computes 32x64 = 2x4 WMMA tiles.
// MODE 0: store bf16 to outb (ldc, +coloff)
// MODE 1: gate epilogue: outb = bf16( sigmoid(acc + bias[col]) * gsrc[row,col] )
// MODE 2: residual epilogue: outf = acc + resid[row,col]  (fp32)
// All M,N multiples of 128 and K multiple of 32 in this pipeline -> no guards.
// ---------------------------------------------------------------------------
template <int MODE>
__global__ __launch_bounds__(256) void gemm_wmma(
    const bf16* __restrict__ A, const bf16* __restrict__ W,
    int K, int lda, int ldb,
    bf16* __restrict__ outb, int ldc, int coloff,
    const float* __restrict__ bias, const bf16* __restrict__ gsrc,
    const float* __restrict__ resid, float* __restrict__ outf, int ldcf) {
  __shared__ bf16 sA[2][128][32];
  __shared__ bf16 sB[2][128][32];

  const int lane = threadIdx.x & 31;
  const int wv   = threadIdx.x >> 5;
  const int wm   = wv >> 1;     // 0..3
  const int wn   = wv & 1;      // 0..1
  const int l15  = lane & 15;
  const int ph   = lane >> 4;   // lane half (ISA fragment layout)
  const int bm = blockIdx.y * 128;
  const int bn = blockIdx.x * 128;

  // async staging of one 128x32 K-tile of A and of W into LDS buffer `buf`
  auto stage = [&](int buf, int kk) {
#pragma unroll
    for (int s = 0; s < 2; ++s) {
      const int c   = threadIdx.x + s * 256;  // 512 x 16B chunks per operand
      const int row = c >> 2;
      const int ko  = (c & 3) * 8;
      async_cp16(A + (size_t)(bm + row) * lda + kk + ko, &sA[buf][row][ko]);
      async_cp16(W + (size_t)(bn + row) * ldb + kk + ko, &sB[buf][row][ko]);
    }
  };

  v8f acc[2][4];
  {
    v8f z = {};
#pragma unroll
    for (int i = 0; i < 2; ++i)
#pragma unroll
      for (int j = 0; j < 4; ++j) acc[i][j] = z;
  }

  stage(0, 0);
  for (int k0 = 0; k0 < K; k0 += 32) {
    const int buf = (k0 >> 5) & 1;
    wait_asynccnt0();      // this wave's copies into `buf` complete
    __syncthreads();       // all threads' copies visible
    if (k0 + 32 < K) {
      stage(buf ^ 1, k0 + 32);  // overlap next tile with compute
      // L2 warm-up for the tile after next (global_prefetch_b8)
      if (k0 + 64 < K) {
        __builtin_prefetch(A + (size_t)(bm + l15 * 8) * lda + k0 + 64, 0, 1);
        __builtin_prefetch(W + (size_t)(bn + l15 * 8) * ldb + k0 + 64, 0, 1);
      }
    }

    v16bf af[2], bfg[4];
#pragma unroll
    for (int mi = 0; mi < 2; ++mi) {
      // A frag 16x32: lane half ph -> K in [ph*8, ph*8+8) and [16+ph*8, ...)
      const bf16* p = &sA[buf][wm * 32 + mi * 16 + l15][ph * 8];
      af[mi] = pack16(*(const v8bf*)p, *(const v8bf*)(p + 16));
    }
#pragma unroll
    for (int ni = 0; ni < 4; ++ni) {
      // B frag 32x16 from W rows (W^T columns): K in [ph*8, ph*8+8) and [16+ph*8, ...)
      const bf16* p = &sB[buf][wn * 64 + ni * 16 + l15][ph * 8];
      bfg[ni] = pack16(*(const v8bf*)p, *(const v8bf*)(p + 16));
    }
#pragma unroll
    for (int mi = 0; mi < 2; ++mi)
#pragma unroll
      for (int ni = 0; ni < 4; ++ni)
        acc[mi][ni] = __builtin_amdgcn_wmma_f32_16x16x32_bf16(
            false, af[mi], false, bfg[ni], (short)0, acc[mi][ni], false, false);
    // no trailing barrier: buffer `buf` is only overwritten after the next top barrier
  }

#pragma unroll
  for (int mi = 0; mi < 2; ++mi) {
    const int row0 = bm + wm * 32 + mi * 16 + ph * 8;  // C: VGPR r -> M = 8*half + r
#pragma unroll
    for (int ni = 0; ni < 4; ++ni) {
      const int col = bn + wn * 64 + ni * 16 + l15;
#pragma unroll
      for (int r = 0; r < 8; ++r) {
        const float v  = acc[mi][ni][r];
        const int  row = row0 + r;
        if (MODE == 0) {
          outb[(size_t)row * ldc + coloff + col] = (bf16)v;
        } else if (MODE == 1) {
          float g = 1.0f / (1.0f + __expf(-(v + bias[col])));
          float c = (float)gsrc[(size_t)row * ldc + col];
          outb[(size_t)row * ldc + col] = (bf16)(g * c);
        } else {
          outf[(size_t)row * ldcf + col] = v + resid[(size_t)row * ldcf + col];
        }
      }
    }
  }
}

// ---------------------------------------------------------------------------
// Directional masked attention, flash-style online softmax, WMMA everywhere.
// qkv: [B*T, 3072] bf16, cols = {q,k,v} x head(16) x hd(64).
// Block: 256 thr = 8 waves; block owns 128 q-rows, wave owns 16 q-rows.
// Loop over K/V in chunks of 32 columns staged in LDS (V transposed so
// P.V B-fragments are contiguous).  dir: 0=lr 1=rl 2=td 3=bu (index predicate).
// ---------------------------------------------------------------------------
__global__ __launch_bounds__(256) void attn_wmma(const bf16* __restrict__ qkv,
                                                 bf16* __restrict__ y, int dir) {
  __shared__ bf16 sK[32][64];       // K chunk, row-major
  __shared__ bf16 sVt[64][40];      // V chunk transposed (+pad, 16B-aligned rows)
  __shared__ bf16 sP[8][16][32];    // per-wave P staging (C-layout -> A-frag)

  const int lane = threadIdx.x & 31;
  const int wv   = threadIdx.x >> 5;
  const int l15  = lane & 15;
  const int ph   = lane >> 4;
  const int bb   = blockIdx.y >> 4;
  const int hh   = blockIdx.y & 15;
  const int q0   = blockIdx.x * 128;
  const int m0   = q0 + wv * 16;
  const size_t ld = 3072;
  const bf16* qp = qkv + (size_t)bb * 1024 * ld + hh * 64;
  const bf16* kp = qp + 1024;
  const bf16* vp = qp + 2048;

  // Q fragments (16x64 split into two 16x32 A-frags), kept resident
  v16bf qf[2];
#pragma unroll
  for (int kb = 0; kb < 2; ++kb) {
    const bf16* p = qp + (size_t)(m0 + l15) * ld + kb * 32 + ph * 8;
    qf[kb] = pack16(*(const v8bf*)p, *(const v8bf*)(p + 16));
  }

  float mrow[8], lrow[8];
#pragma unroll
  for (int r = 0; r < 8; ++r) { mrow[r] = -1e30f; lrow[r] = 0.0f; }
  v8f yacc[4];
  {
    v8f z = {};
#pragma unroll
    for (int i = 0; i < 4; ++i) yacc[i] = z;
  }

  for (int kt = 0; kt < 1024; kt += 32) {
    __syncthreads();
    {  // cooperative K/V chunk load: 256 threads x 8 elems each
      const int t  = threadIdx.x;
      const int kr = t >> 3;
      const int c8 = (t & 7) * 8;
      v8bf kv = *(const v8bf*)(kp + (size_t)(kt + kr) * ld + c8);
      *(v8bf*)&sK[kr][c8] = kv;
      v8bf vvv = *(const v8bf*)(vp + (size_t)(kt + kr) * ld + c8);
#pragma unroll
      for (int i = 0; i < 8; ++i) sVt[c8 + i][kr] = vvv[i];
    }
    __syncthreads();

    // S = Q @ K^T for two 16x16 tiles (chunk cols 0-15, 16-31)
    v8f s0 = {}, s1 = {};
#pragma unroll
    for (int kb = 0; kb < 2; ++kb) {
      const bf16* p0 = &sK[l15][kb * 32 + ph * 8];
      const bf16* p1 = &sK[16 + l15][kb * 32 + ph * 8];
      v16bf b0 = pack16(*(const v8bf*)p0, *(const v8bf*)(p0 + 16));
      v16bf b1 = pack16(*(const v8bf*)p1, *(const v8bf*)(p1 + 16));
      s0 = __builtin_amdgcn_wmma_f32_16x16x32_bf16(false, qf[kb], false, b0, (short)0, s0, false, false);
      s1 = __builtin_amdgcn_wmma_f32_16x16x32_bf16(false, qf[kb], false, b1, (short)0, s1, false, false);
    }

    // online softmax per q-row (rows live across the 16 lanes of a half)
#pragma unroll
    for (int r = 0; r < 8; ++r) {
      const int qi = m0 + ph * 8 + r;
      const int ti = ((qi & 31) << 5) | (qi >> 5);  // transposed grid index
      float sv0 = s0[r] * 0.125f;                   // 1/sqrt(64)
      float sv1 = s1[r] * 0.125f;
      const int j0 = kt + l15, j1 = kt + 16 + l15;
      bool keep0, keep1;
      if (dir == 0)      { keep0 = j0 <= qi; keep1 = j1 <= qi; }
      else if (dir == 1) { keep0 = j0 >= qi; keep1 = j1 >= qi; }
      else {
        const int tj0 = ((j0 & 31) << 5) | (j0 >> 5);
        const int tj1 = ((j1 & 31) << 5) | (j1 >> 5);
        if (dir == 2) { keep0 = tj0 <= ti; keep1 = tj1 <= ti; }
        else          { keep0 = tj0 >= ti; keep1 = tj1 >= ti; }
      }
      sv0 = keep0 ? sv0 : -1e30f;
      sv1 = keep1 ? sv1 : -1e30f;
      float cm = fmaxf(sv0, sv1);
#pragma unroll
      for (int m = 1; m < 16; m <<= 1) cm = fmaxf(cm, __shfl_xor(cm, m, 32));
      const float mn = fmaxf(mrow[r], cm);
      const float al = __expf(mrow[r] - mn);
      const float p0 = __expf(sv0 - mn);
      const float p1 = __expf(sv1 - mn);
      float ps = p0 + p1;
#pragma unroll
      for (int m = 1; m < 16; m <<= 1) ps += __shfl_xor(ps, m, 32);
      lrow[r] = lrow[r] * al + ps;
      mrow[r] = mn;
#pragma unroll
      for (int n4 = 0; n4 < 4; ++n4) yacc[n4][r] *= al;
      sP[wv][ph * 8 + r][l15]      = (bf16)p0;   // C-layout -> LDS
      sP[wv][ph * 8 + r][16 + l15] = (bf16)p1;
    }

    // P (16x32) as A-frag from wave-private LDS (no barrier: same wave)
    v16bf pf;
    {
      const bf16* pa = &sP[wv][l15][ph * 8];
      const bf16* pb = &sP[wv][l15][16 + ph * 8];
      pf = pack16(*(const v8bf*)pa, *(const v8bf*)pb);
    }
    // Y += P @ V  (4 N-tiles of 16 over hd=64)
#pragma unroll
    for (int n4 = 0; n4 < 4; ++n4) {
      const bf16* p = &sVt[n4 * 16 + l15][ph * 8];
      v16bf vf = pack16(*(const v8bf*)p, *(const v8bf*)(p + 16));
      yacc[n4] = __builtin_amdgcn_wmma_f32_16x16x32_bf16(false, pf, false, vf, (short)0, yacc[n4], false, false);
    }
  }

  // normalize and store attention output (bf16 for the projection GEMM)
#pragma unroll
  for (int r = 0; r < 8; ++r) {
    const float inv = 1.0f / lrow[r];
    const int row = bb * 1024 + m0 + ph * 8 + r;
#pragma unroll
    for (int n4 = 0; n4 < 4; ++n4)
      y[(size_t)row * 1024 + hh * 64 + n4 * 16 + l15] = (bf16)(yacc[n4][r] * inv);
  }
}

// ---------------------------------------------------------------------------
// RMSNorm over D=1024 per row (input pre = fused + x in fp32), out fp32
// ---------------------------------------------------------------------------
__global__ __launch_bounds__(256) void rmsnorm_kernel(const float* __restrict__ pre,
                                                      const float* __restrict__ w,
                                                      float* __restrict__ out) {
  __shared__ float red[8];
  const int row = blockIdx.x;
  const int t = threadIdx.x;
  const float4 x4 = ((const float4*)(pre + (size_t)row * 1024))[t];
  float ss = x4.x * x4.x + x4.y * x4.y + x4.z * x4.z + x4.w * x4.w;
#pragma unroll
  for (int m = 1; m < 32; m <<= 1) ss += __shfl_xor(ss, m, 32);
  if ((t & 31) == 0) red[t >> 5] = ss;
  __syncthreads();
  float tot = 0.0f;
#pragma unroll
  for (int i = 0; i < 8; ++i) tot += red[i];
  const float scale = rsqrtf(tot * (1.0f / 1024.0f) + 1e-6f);
  const float4 w4 = ((const float4*)w)[t];
  float4 o;
  o.x = x4.x * scale * w4.x; o.y = x4.y * scale * w4.y;
  o.z = x4.z * scale * w4.z; o.w = x4.w * scale * w4.w;
  ((float4*)(out + (size_t)row * 1024))[t] = o;
}

// ---------------------------------------------------------------------------
// Host-side orchestration
// ---------------------------------------------------------------------------
extern "C" void kernel_launch(void* const* d_in, const int* in_sizes, int n_in,
                              void* d_out, int out_size, void* d_ws, size_t ws_size,
                              hipStream_t stream) {
  (void)in_sizes; (void)n_in; (void)out_size; (void)ws_size;
  constexpr int D  = 1024;
  constexpr int MT = 4096;  // B*T = 4*1024

  const float* x     = (const float*)d_in[0];
  const float* wqkv[4] = {(const float*)d_in[1], (const float*)d_in[3],
                          (const float*)d_in[5], (const float*)d_in[7]};
  const float* wo[4]   = {(const float*)d_in[2], (const float*)d_in[4],
                          (const float*)d_in[6], (const float*)d_in[8]};
  const float* wgate = (const float*)d_in[9];
  const float* bgate = (const float*)d_in[10];
  const float* wout  = (const float*)d_in[11];
  const float* normw = (const float*)d_in[12];
  // d_in[13], d_in[14] = grid_h, grid_w (32x32, baked into mask predicates)

  // workspace carve (256B aligned)
  char* ws = (char*)d_ws;
  size_t off = 0;
  auto carve = [&](size_t bytes) -> void* {
    void* p = ws + off;
    off += (bytes + 255) & ~(size_t)255;
    return p;
  };
  bf16* xb      = (bf16*)carve((size_t)MT * D * 2);
  bf16* wqkvb[4], *wob[4];
  for (int d = 0; d < 4; ++d) wqkvb[d] = (bf16*)carve((size_t)3 * D * D * 2);
  for (int d = 0; d < 4; ++d) wob[d]   = (bf16*)carve((size_t)D * D * 2);
  bf16* wgateb  = (bf16*)carve((size_t)4 * D * 4 * D * 2);
  bf16* woutb   = (bf16*)carve((size_t)D * 4 * D * 2);
  bf16* qkvbuf  = (bf16*)carve((size_t)MT * 3 * D * 2);
  bf16* ybuf    = (bf16*)carve((size_t)MT * D * 2);
  bf16* concat  = (bf16*)carve((size_t)MT * 4 * D * 2);
  bf16* fusedin = (bf16*)carve((size_t)MT * 4 * D * 2);
  float* pre    = (float*)carve((size_t)MT * D * 4);

  auto cvt = [&](const float* s, bf16* dst, size_t n) {
    long n4 = (long)(n / 4);
    long blocks = (n4 + 255) / 256;
    if (blocks > 4096) blocks = 4096;
    cvt_f32_bf16<<<dim3((unsigned)blocks), dim3(256), 0, stream>>>(s, dst, n4);
  };

  cvt(x, xb, (size_t)MT * D);
  for (int d = 0; d < 4; ++d) {
    cvt(wqkv[d], wqkvb[d], (size_t)3 * D * D);
    cvt(wo[d],   wob[d],   (size_t)D * D);
  }
  cvt(wgate, wgateb, (size_t)16 * D * D);
  cvt(wout,  woutb,  (size_t)4 * D * D);

  for (int dir = 0; dir < 4; ++dir) {
    // QKV projection: [4096,1024] @ [1024,3072] -> qkvbuf bf16
    dim3 g1(3072 / 128, MT / 128);
    gemm_wmma<0><<<g1, 256, 0, stream>>>(xb, wqkvb[dir], D, D, D,
                                         qkvbuf, 3072, 0,
                                         nullptr, nullptr, nullptr, nullptr, 0);
    // masked attention per (b, head)
    attn_wmma<<<dim3(8, 64), 256, 0, stream>>>(qkvbuf, ybuf, dir);
    // output projection into concat columns [dir*1024, dir*1024+1024)
    dim3 g2(D / 128, MT / 128);
    gemm_wmma<0><<<g2, 256, 0, stream>>>(ybuf, wob[dir], D, D, D,
                                         concat, 4 * D, dir * D,
                                         nullptr, nullptr, nullptr, nullptr, 0);
  }

  // gate GEMM + sigmoid + elementwise multiply with concat -> fusedin bf16
  dim3 g3(4 * D / 128, MT / 128);
  gemm_wmma<1><<<g3, 256, 0, stream>>>(concat, wgateb, 4 * D, 4 * D, 4 * D,
                                       fusedin, 4 * D, 0,
                                       bgate, concat, nullptr, nullptr, 0);

  // out GEMM + residual add (fp32) -> pre
  dim3 g4(D / 128, MT / 128);
  gemm_wmma<2><<<g4, 256, 0, stream>>>(fusedin, woutb, 4 * D, 4 * D, 4 * D,
                                       nullptr, 0, 0,
                                       nullptr, nullptr, x, pre, D);

  // RMSNorm -> fp32 output
  rmsnorm_kernel<<<dim3(MT), 256, 0, stream>>>(pre, normw, (float*)d_out);
}
